// MultiHeadAttention_58926951301286
// MI455X (gfx1250) — matrix operand order repaired and checked by reading, not compiled, not run on hardware
//
#include <hip/hip_runtime.h>

typedef __attribute__((ext_vector_type(16))) __bf16 v16bf;
typedef __attribute__((ext_vector_type(8)))  float  v8f;
typedef int v4i_vs __attribute__((vector_size(16)));   // matches async-LDS builtin

#define D_MODEL 1024
#define SEQ     2048
#define BATCH   4
#define NHEAD   16
#define HDIM    64

#define GLOBAL_AS __attribute__((address_space(1)))
#define LDS_AS    __attribute__((address_space(3)))

#if defined(__has_builtin)
#  if __has_builtin(__builtin_amdgcn_global_load_async_to_lds_b128)
#    define HAVE_ASYNC_LDS 1
#  endif
#endif

static __device__ inline void wait_async0() {
#if defined(__has_builtin) && __has_builtin(__builtin_amdgcn_s_wait_asynccnt)
  __builtin_amdgcn_s_wait_asynccnt(0);
#else
  asm volatile("s_wait_asynccnt 0" ::: "memory");
#endif
}

// 16B global -> LDS copy; async (ASYNCcnt-tracked) when available.
static __device__ inline void copy_b128_to_lds(__bf16* lds_dst, const __bf16* gsrc) {
#ifdef HAVE_ASYNC_LDS
  __builtin_amdgcn_global_load_async_to_lds_b128(
      (GLOBAL_AS v4i_vs*)gsrc, (LDS_AS v4i_vs*)lds_dst, 0, 0);
#else
  *(uint4*)lds_dst = *(const uint4*)gsrc;
#endif
}

// ---------------------------------------------------------------------------
// WMMA fragment loaders (CDNA5 wave32 layouts, CDNA5 ISA §7.12.2)
//
// A (16x32, 16-bit): lane l holds row m = l&15; elements 0..7 -> k = kb+e,
// elements 8..15 -> k = 16+kb+(e-8), where kb = (l>>4)*8.
// B (32x16, 16-bit): lane l holds col n = l&15; same k mapping as A (kb = (l>>4)*8).
// C/D (16x16 f32): vgpr i, lane l -> row = i + 8*(l>>4), col = l&15.
//
// All fragment loads are CONTIGUOUS per lane (16B chunks -> ds_load_b128)
// because B-operands are staged transposed in LDS.
// ---------------------------------------------------------------------------

static __device__ inline int lane_id() { return (int)(threadIdx.x & 31u); }

// A fragment from a row-major [16 x >=32] tile: two contiguous 16B reads.
static __device__ inline v16bf load_a_frag(const __bf16* base, int stride) {
  const int l = lane_id();
  const __bf16* p = base + (l & 15) * stride + ((l >> 4) << 3);
  v16bf f;
#pragma unroll
  for (int e = 0; e < 8; ++e) { f[e] = p[e]; f[e + 8] = p[e + 16]; }
  return f;
}

// B fragment (logical B[k][n], 32x16) from a TRANSPOSED tile stored [n][k]
// row-major: lane reads k = kb..kb+7 and 16+kb..16+kb+7 (two 16B chunks).
static __device__ inline v16bf load_bT_frag(const __bf16* base, int stride) {
  const int l = lane_id();
  const __bf16* p = base + (l & 15) * stride + ((l >> 4) << 3);
  v16bf f;
#pragma unroll
  for (int e = 0; e < 8; ++e) { f[e] = p[e]; f[e + 8] = p[e + 16]; }
  return f;
}

static __device__ inline v8f wmma_bf16(v16bf a, v16bf b, v8f c) {
  return __builtin_amdgcn_wmma_f32_16x16x32_bf16(false, a, false, b,
                                                 (short)0, c, false, false);
}

// ---------------------------------------------------------------------------
// Projection GEMM: C[M,N] = X[M,K] @ W[K,N] + bias, M=8192, N=K=1024.
// Block tile 128x128, 8 waves (4 down M x 2 across N), each wave 32x64.
// X: fp32 (x_is_bf16=0) or bf16 (x_is_bf16=1, attention output path).
// mode 0: store bf16 scattered to [B,H,S,Hd], scaled (Q gets 1/sqrt(Hd)).
// mode 1: store fp32 row-major [M,N] (final output).
// ---------------------------------------------------------------------------

#define TM   128
#define TN   128
#define TK   32
#define LDA  40    // As: 128 x 32, row-major, padded
#define LDBT 40    // Bs: 128(n) x 32(k), TRANSPOSED, padded

__global__ __launch_bounds__(256)
void proj_gemm_kernel(const void* __restrict__ Xv, int x_is_bf16,
                      const float* __restrict__ W, const float* __restrict__ bias,
                      void* __restrict__ out, int mode, float scale) {
  __shared__ __attribute__((aligned(16))) __bf16 As[TM * LDA];
  __shared__ __attribute__((aligned(16))) __bf16 Bs[TN * LDBT];
  const int bn = blockIdx.x;           // N tile
  const int bm = blockIdx.y;           // M tile
  const int t  = threadIdx.x;
  const int w  = t >> 5;
  const int wm = w & 3;                // wave row (32 rows each)
  const int wn = w >> 2;               // wave col (64 cols each)
  const int l  = t & 31;
  const int grp = l >> 4;
  const int cn  = l & 15;

  v8f zero = {};
  v8f acc[2][4];
#pragma unroll
  for (int i = 0; i < 2; ++i)
#pragma unroll
    for (int j = 0; j < 4; ++j) acc[i][j] = zero;

  const int KT = D_MODEL / TK;
  for (int kt = 0; kt < KT; ++kt) {
    // ---- Stage A tile: 128x32 -> bf16 row-major ----
    if (x_is_bf16) {
      const __bf16* X = (const __bf16*)Xv;
#pragma unroll
      for (int it = 0; it < 2; ++it) {
        const int idx = t + it * 256;
        const int r = idx >> 2, c8 = idx & 3;   // 128 rows x 4 x (8 bf16)
        copy_b128_to_lds(&As[r * LDA + c8 * 8],
                         X + (size_t)(bm * TM + r) * D_MODEL + kt * TK + c8 * 8);
      }
    } else {
      const float* X = (const float*)Xv;
#pragma unroll
      for (int it = 0; it < 4; ++it) {
        const int idx = t + it * 256;
        const int r = idx >> 3, c4 = idx & 7;   // 128 rows x 8 float4 chunks
        const float4 v =
            *(const float4*)(X + (size_t)(bm * TM + r) * D_MODEL + kt * TK + c4 * 4);
        __bf16* dst = &As[r * LDA + c4 * 4];
        dst[0] = (__bf16)v.x; dst[1] = (__bf16)v.y;
        dst[2] = (__bf16)v.z; dst[3] = (__bf16)v.w;
      }
    }
    // ---- Stage B tile: 32x128 fp32 -> bf16, stored TRANSPOSED [n][k] ----
#pragma unroll
    for (int it = 0; it < 4; ++it) {
      const int idx = t + it * 256;
      const int r = idx >> 5, c4 = idx & 31;    // k row, group of 4 n
      const float4 v =
          *(const float4*)(W + (size_t)(kt * TK + r) * D_MODEL + bn * TN + c4 * 4);
      Bs[(c4 * 4 + 0) * LDBT + r] = (__bf16)v.x;
      Bs[(c4 * 4 + 1) * LDBT + r] = (__bf16)v.y;
      Bs[(c4 * 4 + 2) * LDBT + r] = (__bf16)v.z;
      Bs[(c4 * 4 + 3) * LDBT + r] = (__bf16)v.w;
    }
    // Prefetch next k-tile into L2 (global_prefetch_b8).
    if (kt + 1 < KT) {
      const size_t esz = x_is_bf16 ? 2 : 4;
      __builtin_prefetch((const char*)Xv +
          ((size_t)(bm * TM + (t >> 1)) * D_MODEL + (kt + 1) * TK) * esz, 0, 0);
      __builtin_prefetch((const char*)W +
          ((size_t)((kt + 1) * TK + (t >> 3)) * D_MODEL + bn * TN) * 4, 0, 0);
    }
#ifdef HAVE_ASYNC_LDS
    wait_async0();
#endif
    __syncthreads();

    const v16bf a0 = load_a_frag(&As[(wm * 32 + 0) * LDA], LDA);
    const v16bf a1 = load_a_frag(&As[(wm * 32 + 16) * LDA], LDA);
    const v16bf b0 = load_bT_frag(&Bs[(wn * 64 + 0)  * LDBT], LDBT);
    const v16bf b1 = load_bT_frag(&Bs[(wn * 64 + 16) * LDBT], LDBT);
    const v16bf b2 = load_bT_frag(&Bs[(wn * 64 + 32) * LDBT], LDBT);
    const v16bf b3 = load_bT_frag(&Bs[(wn * 64 + 48) * LDBT], LDBT);

    acc[0][0] = wmma_bf16(a0, b0, acc[0][0]);
    acc[0][1] = wmma_bf16(a0, b1, acc[0][1]);
    acc[0][2] = wmma_bf16(a0, b2, acc[0][2]);
    acc[0][3] = wmma_bf16(a0, b3, acc[0][3]);
    acc[1][0] = wmma_bf16(a1, b0, acc[1][0]);
    acc[1][1] = wmma_bf16(a1, b1, acc[1][1]);
    acc[1][2] = wmma_bf16(a1, b2, acc[1][2]);
    acc[1][3] = wmma_bf16(a1, b3, acc[1][3]);
    __syncthreads();
  }

  // Epilogue: bias add + store. Uniform mode branch hoisted out of the loops.
  if (mode == 0) {
#pragma unroll
    for (int i = 0; i < 2; ++i)
#pragma unroll
      for (int j = 0; j < 4; ++j) {
        const int n  = bn * TN + wn * 64 + j * 16 + cn;
        const float bv = bias[n];
        const int h = n >> 6, hd = n & (HDIM - 1);
#pragma unroll
        for (int e = 0; e < 8; ++e) {
          const int m = bm * TM + wm * 32 + i * 16 + e + grp * 8;
          const int bb = m >> 11, s = m & (SEQ - 1);
          ((__bf16*)out)[((((size_t)bb * NHEAD + h) * SEQ) + s) * HDIM + hd] =
              (__bf16)((acc[i][j][e] + bv) * scale);
        }
      }
  } else {
#pragma unroll
    for (int i = 0; i < 2; ++i)
#pragma unroll
      for (int j = 0; j < 4; ++j) {
        const int n  = bn * TN + wn * 64 + j * 16 + cn;
        const float bv = bias[n];
#pragma unroll
        for (int e = 0; e < 8; ++e) {
          const int m = bm * TM + wm * 32 + i * 16 + e + grp * 8;
          ((float*)out)[(size_t)m * D_MODEL + n] = acc[i][j][e] + bv;
        }
      }
  }
}

// ---------------------------------------------------------------------------
// Flash attention: one block = 128 queries of one (b,h); 8 waves x 16 queries.
// K staged [key][hd] via async-to-LDS; V staged TRANSPOSED [hd][key] so P@V
// B-fragments are contiguous. Online softmax in registers. Q pre-scaled by
// 1/sqrt(Hd). Output written bf16 [B,S,D] (consumed by bf16-input final GEMM).
// ---------------------------------------------------------------------------

#define LDK  80   // Ks: 32 x 64, padded to 80 (160B rows, 16B aligned)
#define LDVT 40   // Vt: 64(hd) x 32(key), transposed, padded
#define LDP  40   // per-wave 16 x 32 P tile, padded

__global__ __launch_bounds__(256)
void flash_attn_kernel(const __bf16* __restrict__ Qb, const __bf16* __restrict__ Kb,
                       const __bf16* __restrict__ Vb, __bf16* __restrict__ O) {
  __shared__ __attribute__((aligned(16))) __bf16 Ks[32 * LDK];
  __shared__ __attribute__((aligned(16))) __bf16 Vt[HDIM * LDVT];
  __shared__ __attribute__((aligned(16))) __bf16 Ps[8 * 16 * LDP];

  const int bh = blockIdx.y;           // b*NHEAD + h
  const int qt = blockIdx.x;           // tile of 128 queries
  const int t  = threadIdx.x;
  const int w  = t >> 5;
  const int l  = t & 31;
  const int grp = l >> 4;
  const int cn  = l & 15;

  const __bf16* Qbh = Qb + (size_t)bh * SEQ * HDIM;
  const __bf16* Kbh = Kb + (size_t)bh * SEQ * HDIM;
  const __bf16* Vbh = Vb + (size_t)bh * SEQ * HDIM;

  const int q0 = qt * 128 + w * 16;

  // Q A-fragments for head-dim chunks [0,32) and [32,64).
  v16bf qa[2];
#pragma unroll
  for (int c = 0; c < 2; ++c) {
    const __bf16* p = Qbh + (size_t)(q0 + (l & 15)) * HDIM + c * 32 + ((l >> 4) << 3);
    v16bf f;
#pragma unroll
    for (int e = 0; e < 8; ++e) { f[e] = p[e]; f[e + 8] = p[e + 16]; }
    qa[c] = f;
  }

  v8f zero = {};
  v8f o_acc[4];
#pragma unroll
  for (int j = 0; j < 4; ++j) o_acc[j] = zero;
  float mrow[8], lrow[8];
#pragma unroll
  for (int e = 0; e < 8; ++e) { mrow[e] = -1.0e30f; lrow[e] = 0.0f; }

  __bf16* Pw = &Ps[w * 16 * LDP];

  for (int ktile = 0; ktile < SEQ / 32; ++ktile) {
    // Cooperative stage: K rows copied verbatim (async DMA when available),
    // V transposed to [hd][key] through registers.
    {
      const int r = t >> 3, ch = t & 7;   // key row, 16B chunk (8 bf16)
      copy_b128_to_lds(&Ks[r * LDK + ch * 8],
                       Kbh + (size_t)(ktile * 32 + r) * HDIM + ch * 8);
      const uint4 vv =
          ((const uint4*)(Vbh + (size_t)(ktile * 32 + r) * HDIM))[ch];
      const __bf16* pv = (const __bf16*)&vv;
#pragma unroll
      for (int j = 0; j < 8; ++j) Vt[(ch * 8 + j) * LDVT + r] = pv[j];
    }
    if (ktile + 1 < SEQ / 32) {   // prefetch next KV tile into L2
      __builtin_prefetch(Kbh + (size_t)((ktile + 1) * 32 + (t >> 3)) * HDIM, 0, 0);
      __builtin_prefetch(Vbh + (size_t)((ktile + 1) * 32 + (t >> 3)) * HDIM, 0, 0);
    }
#ifdef HAVE_ASYNC_LDS
    wait_async0();
#endif
    __syncthreads();

    // Scores: two 16x16 tiles (keys 0-15 and 16-31), k-dim = 64 over 2 WMMAs.
    // Logical B[k=hd][n=key] = Ks[key][hd] (stored [n][k] -> load_bT_frag).
    v8f s0 = zero, s1 = zero;
    {
      const v16bf k00 = load_bT_frag(&Ks[0 * LDK + 0],   LDK);
      const v16bf k01 = load_bT_frag(&Ks[0 * LDK + 32],  LDK);
      const v16bf k10 = load_bT_frag(&Ks[16 * LDK + 0],  LDK);
      const v16bf k11 = load_bT_frag(&Ks[16 * LDK + 32], LDK);
      s0 = wmma_bf16(qa[0], k00, s0);
      s0 = wmma_bf16(qa[1], k01, s0);
      s1 = wmma_bf16(qa[0], k10, s1);
      s1 = wmma_bf16(qa[1], k11, s1);
    }

    // Online softmax. Row r = e + 8*grp lives on the 16 lanes of this group.
    float tmax[8];
#pragma unroll
    for (int e = 0; e < 8; ++e) tmax[e] = fmaxf(s0[e], s1[e]);
#pragma unroll
    for (int mk = 1; mk <= 8; mk <<= 1)
#pragma unroll
      for (int e = 0; e < 8; ++e)
        tmax[e] = fmaxf(tmax[e], __shfl_xor(tmax[e], mk, 32));

    float alpha[8];
#pragma unroll
    for (int e = 0; e < 8; ++e) {
      const float nm = fmaxf(mrow[e], tmax[e]);
      alpha[e] = __expf(mrow[e] - nm);
      mrow[e] = nm;
    }
    v8f p0 = zero, p1 = zero;
    float rsum[8];
#pragma unroll
    for (int e = 0; e < 8; ++e) {
      p0[e] = __expf(s0[e] - mrow[e]);
      p1[e] = __expf(s1[e] - mrow[e]);
      rsum[e] = p0[e] + p1[e];
    }
#pragma unroll
    for (int mk = 1; mk <= 8; mk <<= 1)
#pragma unroll
      for (int e = 0; e < 8; ++e) rsum[e] += __shfl_xor(rsum[e], mk, 32);
#pragma unroll
    for (int e = 0; e < 8; ++e) lrow[e] = lrow[e] * alpha[e] + rsum[e];
#pragma unroll
    for (int j = 0; j < 4; ++j)
#pragma unroll
      for (int e = 0; e < 8; ++e) o_acc[j][e] *= alpha[e];

    // C-layout -> A-layout for P via this wave's private LDS slab.
#pragma unroll
    for (int e = 0; e < 8; ++e) {
      Pw[(e + grp * 8) * LDP + cn]      = (__bf16)p0[e];
      Pw[(e + grp * 8) * LDP + 16 + cn] = (__bf16)p1[e];
    }
    asm volatile("s_wait_dscnt 0" ::: "memory");  // same-wave LDS RAW
    const v16bf pf = load_a_frag(Pw, LDP);

    // P @ V: logical B[k=key][n=hd] = Vt[hd][key] (stored [n][k]).
    {
      const v16bf vf0 = load_bT_frag(&Vt[(0)  * LDVT], LDVT);
      const v16bf vf1 = load_bT_frag(&Vt[(16) * LDVT], LDVT);
      const v16bf vf2 = load_bT_frag(&Vt[(32) * LDVT], LDVT);
      const v16bf vf3 = load_bT_frag(&Vt[(48) * LDVT], LDVT);
      o_acc[0] = wmma_bf16(pf, vf0, o_acc[0]);
      o_acc[1] = wmma_bf16(pf, vf1, o_acc[1]);
      o_acc[2] = wmma_bf16(pf, vf2, o_acc[2]);
      o_acc[3] = wmma_bf16(pf, vf3, o_acc[3]);
    }
    __syncthreads();
  }

  // Normalize and store bf16 [B,S,D].
#pragma unroll
  for (int e = 0; e < 8; ++e) lrow[e] = 1.0f / lrow[e];
  const int b = bh >> 4, h = bh & 15;
#pragma unroll
  for (int j = 0; j < 4; ++j)
#pragma unroll
    for (int e = 0; e < 8; ++e) {
      const int s = q0 + e + grp * 8;
      O[((size_t)b * SEQ + s) * D_MODEL + h * HDIM + j * 16 + cn] =
          (__bf16)(o_acc[j][e] * lrow[e]);
    }
}

// ---------------------------------------------------------------------------
// Launch: 3 projections -> flash attention -> output projection.
// Workspace: Qb/Kb/Vb bf16 [B,H,S,Hd] + Ob bf16 [B,S,D] (16 MB each, 64 MB).
// ---------------------------------------------------------------------------

extern "C" void kernel_launch(void* const* d_in, const int* in_sizes, int n_in,
                              void* d_out, int out_size, void* d_ws, size_t ws_size,
                              hipStream_t stream) {
  (void)in_sizes; (void)n_in; (void)out_size; (void)ws_size;
  const float* x  = (const float*)d_in[0];
  const float* Wq = (const float*)d_in[1];
  const float* bq = (const float*)d_in[2];
  const float* Wk = (const float*)d_in[3];
  const float* bk = (const float*)d_in[4];
  const float* Wv = (const float*)d_in[5];
  const float* bv = (const float*)d_in[6];
  const float* Wo = (const float*)d_in[7];
  const float* bo = (const float*)d_in[8];

  const size_t qkv_elems = (size_t)BATCH * NHEAD * SEQ * HDIM;  // 8M
  __bf16* Qb = (__bf16*)d_ws;
  __bf16* Kb = Qb + qkv_elems;
  __bf16* Vb = Kb + qkv_elems;
  __bf16* Ob = Vb + qkv_elems;

  const dim3 blk(256);
  const dim3 gGemm(D_MODEL / TN, (BATCH * SEQ) / TM);  // (8, 64)
  const float qscale = 0.125f;                         // 1/sqrt(64)

  hipLaunchKernelGGL(proj_gemm_kernel, gGemm, blk, 0, stream,
                     (const void*)x, 0, Wq, bq, (void*)Qb, 0, qscale);
  hipLaunchKernelGGL(proj_gemm_kernel, gGemm, blk, 0, stream,
                     (const void*)x, 0, Wk, bk, (void*)Kb, 0, 1.0f);
  hipLaunchKernelGGL(proj_gemm_kernel, gGemm, blk, 0, stream,
                     (const void*)x, 0, Wv, bv, (void*)Vb, 0, 1.0f);

  const dim3 gAttn(SEQ / 128, BATCH * NHEAD);          // (16, 64)
  hipLaunchKernelGGL(flash_attn_kernel, gAttn, blk, 0, stream, Qb, Kb, Vb, Ob);

  hipLaunchKernelGGL(proj_gemm_kernel, gGemm, blk, 0, stream,
                     (const void*)Ob, 1, Wo, bo, d_out, 1, 1.0f);
}
